// MilliesRNN_89601607729293
// MI455X (gfx1250) — matrix operand order repaired and checked by reading, not compiled, not run on hardware
//
#include <hip/hip_runtime.h>
#include <hip/hip_bf16.h>

typedef __bf16 bf16;
typedef __attribute__((ext_vector_type(16))) __bf16 v16bf;
typedef __attribute__((ext_vector_type(8)))  __bf16 v8bf;
typedef __attribute__((ext_vector_type(8)))  float  v8f;
typedef __attribute__((ext_vector_type(4)))  int    v4i;

#define AS1 __attribute__((address_space(1)))
#define AS3 __attribute__((address_space(3)))

#define B_ 64
#define T_ 2048
#define I_ 256
#define H_ 512
#define O_ 512

#define LDH    (H_ + 8)   // padded LDS row stride in halves (1040 B, 16B aligned,
                          // 260 dwords == 4 mod 64 banks -> conflict-free)
#define NCHUNK 128        // output columns per scan workgroup (N split)
#define NGRP   4          // workgroups cooperating per batch group

// ---------------------------------------------------------------------------
// WMMA helpers (CDNA5 16x16x32 bf16 -> f32 accumulate)
// ---------------------------------------------------------------------------
__device__ __forceinline__ v8f wmma_bf16(v16bf a, v16bf b, v8f c) {
    // (neg_a, A, neg_b, B, c_mod, C, reuse_a, reuse_b)
    return __builtin_amdgcn_wmma_f32_16x16x32_bf16(
        false, a, false, b, (short)0, c, false, false);
}

// A-matrix 16x32 (MxK) bf16 fragment from row-major storage.
// lanes 0-15: row M=lane, halves 0..7 = K 0..7, halves 8..15 = K 16..23;
// lanes 16-31: row M=lane-16, K ranges shifted by +8.
template <int STRIDE>
__device__ __forceinline__ v16bf load_frag_a(const bf16* base, int lane) {
    const bf16* p = base + (lane & 15) * STRIDE + ((lane >> 4) << 3);
    v8bf lo = *(const v8bf*)(p);
    v8bf hi = *(const v8bf*)(p + 16);
    return __builtin_shufflevector(lo, hi, 0, 1, 2, 3, 4, 5, 6, 7,
                                           8, 9, 10, 11, 12, 13, 14, 15);
}

// B-matrix 32x16 (KxN) bf16 fragment, B[k][n] = W[n][k] (W row-major, ld=ldw).
// lanes 0-15: column n=lane, halves 0..7 = K 0..7, halves 8..15 = K 16..23;
// lanes 16-31: K ranges shifted by +8.
__device__ __forceinline__ v16bf load_frag_b(const bf16* w, int ldw, int lane) {
    const bf16* p = w + (size_t)(lane & 15) * ldw + ((lane >> 4) << 3);
    v8bf lo = *(const v8bf*)(p);
    v8bf hi = *(const v8bf*)(p + 16);
    return __builtin_shufflevector(lo, hi, 0, 1, 2, 3, 4, 5, 6, 7,
                                           8, 9, 10, 11, 12, 13, 14, 15);
}

__device__ __forceinline__ float retanh(float x) {
    return tanhf(fmaxf(x, 0.0f));
}

// ---------------------------------------------------------------------------
// CDNA5 async global->LDS copy (ASYNCcnt-tracked), 16 B per lane.
// Builtin signature (per hipcc diagnostic): param0 = v4i in global AS(1),
// param1 = LDS destination, plus imm offset / cpol.
// ---------------------------------------------------------------------------
__device__ __forceinline__ void async_copy_g2l_b128(const bf16* gsrc, bf16* ldst) {
#if defined(__has_builtin) && __has_builtin(__builtin_amdgcn_global_load_async_to_lds_b128)
    __builtin_amdgcn_global_load_async_to_lds_b128(
        (AS1 v4i*)gsrc, (AS3 v4i*)ldst, 0, 0);
#else
    unsigned loff = (unsigned)(size_t)(AS3 char*)ldst;
    unsigned long long gaddr = (unsigned long long)(size_t)gsrc;
    asm volatile("global_load_async_to_lds_b128 %0, %1, off"
                 :: "v"(loff), "v"(gaddr) : "memory");
#endif
}

__device__ __forceinline__ void wait_async_zero() {
#if defined(__has_builtin) && __has_builtin(__builtin_amdgcn_s_wait_asynccnt)
    __builtin_amdgcn_s_wait_asynccnt(0);
#else
    asm volatile("s_wait_asynccnt 0x0" ::: "memory");
#endif
}

// ---------------------------------------------------------------------------
// Kernel 1: f32 -> bf16 weight conversion + sync-counter reset.
// ---------------------------------------------------------------------------
__global__ __launch_bounds__(256) void cvt_weights(
    const float* __restrict__ Wi, const float* __restrict__ Wh,
    const float* __restrict__ Wo, bf16* __restrict__ wi,
    bf16* __restrict__ wh, bf16* __restrict__ wo, int* __restrict__ cnt) {
    int idx = blockIdx.x * 256 + threadIdx.x;
    if (idx < 16) cnt[idx] = 0;                 // deterministic across replays
    if (idx < H_ * I_) wi[idx] = (bf16)Wi[idx];
    if (idx < H_ * H_) {
        wh[idx] = (bf16)Wh[idx];
        wo[idx] = (bf16)Wo[idx];
    }
}

// ---------------------------------------------------------------------------
// Kernel 2: x_proj = data @ Wi^T + bi  (stored bf16), fully parallel.
// One block per 16 rows of [B*T, I]; 8 waves x 64 output columns each.
// ---------------------------------------------------------------------------
__global__ __launch_bounds__(256) void xproj_gemm(
    const float* __restrict__ data, const bf16* __restrict__ wi,
    const float* __restrict__ bi, bf16* __restrict__ xproj) {
    __shared__ bf16 at[16][I_ + 8];

    const int mt  = blockIdx.x;
    const int tid = threadIdx.x;

    {   // cooperative f32->bf16 stage of the 16x256 A tile
        int r = tid >> 4;
        int c = (tid & 15) * 16;
        const float* src = data + (size_t)(mt * 16 + r) * I_ + c;
#pragma unroll
        for (int j = 0; j < 16; ++j) at[r][c + j] = (bf16)src[j];
    }
    __syncthreads();

    const int wid  = tid >> 5;
    const int lane = tid & 31;
    const int n0   = wid * 64;
    const int hi16 = lane >> 4;
    const int col  = lane & 15;

    v8f acc[4];
#pragma unroll
    for (int t = 0; t < 4; ++t) {
        float bv = bi[n0 + t * 16 + col];
#pragma unroll
        for (int g = 0; g < 8; ++g) acc[t][g] = bv;   // bias folded into C
    }

#pragma unroll
    for (int ks = 0; ks < I_ / 32; ++ks) {
        v16bf a = load_frag_a<I_ + 8>(&at[0][0] + ks * 32, lane);
#pragma unroll
        for (int t = 0; t < 4; ++t) {
            v16bf b = load_frag_b(wi + (size_t)(n0 + t * 16) * I_ + ks * 32, I_, lane);
            acc[t] = wmma_bf16(a, b, acc[t]);
        }
    }

#pragma unroll
    for (int t = 0; t < 4; ++t)
#pragma unroll
        for (int g = 0; g < 8; ++g) {
            int row = mt * 16 + g + 8 * hi16;
            xproj[(size_t)row * H_ + n0 + t * 16 + col] = (bf16)acc[t][g];
        }
}

// ---------------------------------------------------------------------------
// Kernel 3: sequential scan, 16 persistent workgroups:
//   batch group g = blockIdx/4 (16 rows), N chunk c = blockIdx%4 (128 cols).
// Wh/Wo slices (130 KB each) async-preloaded into LDS once; h tile in LDS.
// Cross-WG h exchange via L2 + monotonic arrival counters; the per-step
// h pull into LDS uses the CDNA5 async global->LDS pipe.
// ---------------------------------------------------------------------------
__global__ __launch_bounds__(256) void rnn_scan(
    const bf16* __restrict__ xproj, const bf16* __restrict__ wh,
    const float* __restrict__ bh, const bf16* __restrict__ wo,
    const float* __restrict__ bo, bf16* __restrict__ hx,
    int* __restrict__ cnt, float* __restrict__ outputs,
    float* __restrict__ hiddens) {
    extern __shared__ char smem[];
    bf16* whL = (bf16*)smem;              // [NCHUNK][LDH]  (133,120 B)
    bf16* woL = whL + NCHUNK * LDH;       // [NCHUNK][LDH]
    bf16* hL  = woL + NCHUNK * LDH;       // [16][LDH]

    const int grp   = blockIdx.x >> 2;    // 0..3  batch group
    const int chunk = blockIdx.x & 3;     // 0..3  N chunk
    const int bbase = grp * 16;
    const int tid   = threadIdx.x;
    const int wid   = tid >> 5;           // 0..7
    const int lane  = tid & 31;
    const int n_l   = wid * 16;           // local column tile
    const int n_g   = chunk * NCHUNK + n_l;
    const int hi16  = lane >> 4;
    const int col   = lane & 15;
    const int n     = n_g + col;

    // Async-preload this chunk's Wh/Wo rows into LDS (padded stride LDH).
    for (int i = tid; i < NCHUNK * (H_ / 8); i += 256) {
        int r = i >> 6;               // 64 8-half segments per row
        int c = (i & 63) * 8;
        size_t gsrc = (size_t)(chunk * NCHUNK + r) * H_ + c;
        async_copy_g2l_b128(wh + gsrc, whL + r * LDH + c);
        async_copy_g2l_b128(wo + gsrc, woL + r * LDH + c);
    }
    // h0 = 0
    for (int i = tid; i < 16 * LDH; i += 256) hL[i] = (bf16)0.0f;
    wait_async_zero();
    __syncthreads();

    int* mycnt = cnt + grp;
    const float bhv = bh[n];
    const float bov = bo[n];

    for (int t = 0; t < T_; ++t) {
        // ---- phase 1: h_new = retanh(x_t + h @ Wh^T + bh) ----
        v8f accA, accB;
#pragma unroll
        for (int g = 0; g < 8; ++g) {
            int m = g + 8 * hi16;
            accA[g] = (float)xproj[((size_t)(bbase + m) * T_ + t) * H_ + n] + bhv;
            accB[g] = 0.0f;
        }
        // two independent K-chains (K 0..255 and 256..511) for ILP
#pragma unroll 4
        for (int ks = 0; ks < 8; ++ks) {
            v16bf a0 = load_frag_a<LDH>(hL + ks * 32, lane);
            v16bf b0 = load_frag_b(whL + n_l * LDH + ks * 32, LDH, lane);
            accA = wmma_bf16(a0, b0, accA);
            v16bf a1 = load_frag_a<LDH>(hL + (ks + 8) * 32, lane);
            v16bf b1 = load_frag_b(whL + n_l * LDH + (ks + 8) * 32, LDH, lane);
            accB = wmma_bf16(a1, b1, accB);
        }

        // activation; publish own 16x128 slice to L2 exchange buffer + hiddens
        bf16* hxp = hx + (size_t)((t & 1) * NGRP + grp) * (16 * H_);
#pragma unroll
        for (int g = 0; g < 8; ++g) {
            int m = g + 8 * hi16;
            float v = retanh(accA[g] + accB[g]);
            hiddens[((size_t)(bbase + m) * T_ + t) * H_ + n] = v;
            hxp[m * H_ + n] = (bf16)v;
        }

        // ---- cross-workgroup step barrier (release/acquire via L2) ----
        __threadfence();
        __syncthreads();                       // all waves done reading hL too
        if (tid == 0) {
            __hip_atomic_fetch_add(mycnt, 1, __ATOMIC_RELEASE,
                                   __HIP_MEMORY_SCOPE_AGENT);
            while (__hip_atomic_load(mycnt, __ATOMIC_ACQUIRE,
                                     __HIP_MEMORY_SCOPE_AGENT) < NGRP * (t + 1))
                __builtin_amdgcn_s_sleep(2);
        }
        __syncthreads();
        __threadfence();

        // async-pull the full 16x512 h_new into LDS (L2 -> LDS, no VGPR trip)
        for (int i = tid; i < 16 * (H_ / 8); i += 256) {
            int r = i >> 6;
            int c = (i & 63) * 8;
            async_copy_g2l_b128(hxp + r * H_ + c, hL + r * LDH + c);
        }
        wait_async_zero();
        __syncthreads();

        // ---- phase 2: out = h_new @ Wo^T + bo ----
        v8f ocA, ocB;
#pragma unroll
        for (int g = 0; g < 8; ++g) { ocA[g] = bov; ocB[g] = 0.0f; }
#pragma unroll 4
        for (int ks = 0; ks < 8; ++ks) {
            v16bf a0 = load_frag_a<LDH>(hL + ks * 32, lane);
            v16bf b0 = load_frag_b(woL + n_l * LDH + ks * 32, LDH, lane);
            ocA = wmma_bf16(a0, b0, ocA);
            v16bf a1 = load_frag_a<LDH>(hL + (ks + 8) * 32, lane);
            v16bf b1 = load_frag_b(woL + n_l * LDH + (ks + 8) * 32, LDH, lane);
            ocB = wmma_bf16(a1, b1, ocB);
        }
#pragma unroll
        for (int g = 0; g < 8; ++g) {
            int m = g + 8 * hi16;
            outputs[((size_t)(bbase + m) * T_ + t) * O_ + n] = ocA[g] + ocB[g];
        }
        // no trailing barrier: next phase-1 only reads hL; hL is next written
        // after the next step's counter barrier.
    }
}

// ---------------------------------------------------------------------------
extern "C" void kernel_launch(void* const* d_in, const int* in_sizes, int n_in,
                              void* d_out, int out_size, void* d_ws, size_t ws_size,
                              hipStream_t stream) {
    (void)in_sizes; (void)n_in; (void)out_size; (void)ws_size;
    const float* data = (const float*)d_in[0];
    const float* Wi   = (const float*)d_in[1];
    const float* bi   = (const float*)d_in[2];
    const float* Wh   = (const float*)d_in[3];
    const float* bh   = (const float*)d_in[4];
    const float* Wo   = (const float*)d_in[5];
    const float* bo   = (const float*)d_in[6];

    float* outputs = (float*)d_out;                    // [B,T,O]
    float* hiddens = outputs + (size_t)B_ * T_ * O_;   // [B,T,H]

    char* ws = (char*)d_ws;
    bf16* wi    = (bf16*)(ws);                               // 256 KB
    bf16* wh    = (bf16*)(ws + (1u << 18));                  // 512 KB
    bf16* wo    = (bf16*)(ws + (1u << 18) + (1u << 19));     // 512 KB
    bf16* xproj = (bf16*)(ws + (1u << 18) + (2u << 19));     // 128 MB
    size_t xbytes = (size_t)B_ * T_ * H_ * sizeof(bf16);
    bf16* hx    = (bf16*)(ws + (1u << 18) + (2u << 19) + xbytes);   // 128 KB
    int*  cnt   = (int*)((char*)hx + (size_t)2 * NGRP * 16 * H_ * sizeof(bf16));

    cvt_weights<<<(H_ * H_ + 255) / 256, 256, 0, stream>>>(Wi, Wh, Wo, wi, wh, wo, cnt);
    xproj_gemm<<<(B_ * T_) / 16, 256, 0, stream>>>(data, wi, bi, xproj);

    const int smem = (2 * NCHUNK + 16) * LDH * (int)sizeof(bf16);   // 282,880 B
    rnn_scan<<<B_ / 16 * NGRP, 256, smem, stream>>>(
        xproj, wh, bh, wo, bo, hx, cnt, outputs, hiddens);
}
